// MambaSegmentation_23192823398450
// MI455X (gfx1250) — matrix operand order repaired and checked by reading, not compiled, not run on hardware
//
#include <hip/hip_runtime.h>
#include <cstdint>

typedef __attribute__((ext_vector_type(16))) _Float16 v16h;
typedef __attribute__((ext_vector_type(8)))  _Float16 v8h;
typedef __attribute__((ext_vector_type(8)))  float    v8f;
typedef __attribute__((ext_vector_type(4)))  unsigned int v4u;

union FragA { v8h h8[2]; v16h h16; };
union FragB { v4u u4[2]; v16h h16; };

__device__ __forceinline__ float siluf(float x) { return x / (1.f + __expf(-x)); }
__device__ __forceinline__ float softplusf(float x) { return (x > 20.f) ? x : log1pf(__expf(x)); }

// ---------------------------------------------------------------------------
// Weight pre-pack into per-lane WMMA B-fragment layout (f32 -> f16 dwords).
// mode 0: W is [Nout, K] row-major (GEMM, B = W^T so B(k,n) = W[n*K+k])
// mode 1: W is [Cout, Cin, 3, 3]; linear K enumerated position-major:
//         chunk c = p*(Cin/32)+cb, within-chunk kl -> cin = cb*32+kl, tap p.
// Packed layout: out[((nt*nchunks + c)*32 + lane)*8 + j] holds halves
//   K = c*32 + (lane>>4)*8 + (j&4)*2 + 2j + {0,1},  N = nt*16 + (lane&15).
// ---------------------------------------------------------------------------
__global__ void ms_pack_b(const float* __restrict__ w, uint32_t* __restrict__ out,
                          int total, int nchunks, int K, int Nout, int mode, int Cin) {
  int idx = blockIdx.x * blockDim.x + threadIdx.x;
  if (idx >= total) return;
  int j  = idx & 7;
  int l  = (idx >> 3) & 31;
  int cc = (idx >> 8) % nchunks;
  int nt = (idx >> 8) / nchunks;
  int g  = l >> 4;
  int n  = nt * 16 + (l & 15);
  union { _Float16 h[2]; uint32_t u; } pk;
  for (int hh = 0; hh < 2; ++hh) {
    int kl = g * 8 + (j & 4) * 2 + 2 * j + hh;
    int k  = cc * 32 + kl;
    float v = 0.f;
    if (n < Nout && k < K) {
      if (mode == 0) {
        v = w[(size_t)n * K + k];
      } else {
        int nchpp = Cin >> 5;
        int p   = cc / nchpp;
        int cin = (cc % nchpp) * 32 + kl;
        v = w[((size_t)n * Cin + cin) * 9 + p];
      }
    }
    pk.h[hh] = (_Float16)v;
  }
  out[idx] = pk.u;
}

// ---------------------------------------------------------------------------
// Generic WMMA GEMM: C[M, Npad] = A[M, K] * W^T, f16 WMMA, f32 accumulate.
// Block = 128 threads = 4 waves; wave w -> N-tile blockIdx.y*4+w; M-tile = 16.
// Next-chunk A and B lines are warmed with global_prefetch_b8.
// ---------------------------------------------------------------------------
__global__ void ms_gemm_wmma(const float* __restrict__ A, const uint32_t* __restrict__ Wp,
                             float* __restrict__ C, int K, int Npad) {
  __shared__ alignas(16) _Float16 lA[16 * 32];
  const int tid  = threadIdx.x;
  const int wave = tid >> 5, lane = tid & 31;
  const int m0   = blockIdx.x * 16;
  const int nt   = blockIdx.y * 4 + wave;
  const bool act = (nt * 16) < Npad;           // wave-uniform -> EXEC stays full
  const int nchunks = K >> 5;
  const int srow = tid >> 3;                    // 0..15
  const int skq  = (tid & 7) * 4;               // 0,4,...,28
  v8f acc = {};
  for (int c = 0; c < nchunks; ++c) {
    const float* ap = A + (size_t)(m0 + srow) * K + c * 32 + skq;
    if (c + 1 < nchunks) __builtin_prefetch(ap + 32, 0, 1);   // global_prefetch_b8
    const float4 f = *(const float4*)ap;
    union { _Float16 h[4]; uint64_t u; } pk;
    pk.h[0] = (_Float16)f.x; pk.h[1] = (_Float16)f.y;
    pk.h[2] = (_Float16)f.z; pk.h[3] = (_Float16)f.w;
    *(uint64_t*)(&lA[srow * 32 + skq]) = pk.u;
    __syncthreads();
    if (act) {
      const int ml = lane & 15, g = lane >> 4;
      FragA a;
      a.h8[0] = *(const v8h*)(&lA[ml * 32 + g * 8]);
      a.h8[1] = *(const v8h*)(&lA[ml * 32 + 16 + g * 8]);
      const uint32_t* bp = Wp + (((size_t)nt * nchunks + c) * 32 + lane) * 8;
      if (c + 1 < nchunks) __builtin_prefetch(bp + 256, 0, 1); // next chunk's B frags
      FragB b;
      b.u4[0] = *(const v4u*)bp;
      b.u4[1] = *(const v4u*)(bp + 4);
      acc = __builtin_amdgcn_wmma_f32_16x16x32_f16(false, a.h16, false, b.h16,
                                                   (short)0, acc, false, false);
    }
    __syncthreads();
  }
  if (act) {
    const int g = lane >> 4, n = nt * 16 + (lane & 15);
    for (int v = 0; v < 8; ++v)
      C[(size_t)(m0 + v + g * 8) * Npad + n] = acc[v];
  }
}

// ---------------------------------------------------------------------------
// Implicit-GEMM 3x3 conv (pad=1) + fused BN(eval)+ReLU. Two-pointer input
// supports fused channel concat (cin < split -> in1 else in2).
// Block = 128 threads = 4 waves; wave -> cout tile; 16 consecutive row pixels.
// ---------------------------------------------------------------------------
__global__ void ms_conv3x3_wmma(const float* __restrict__ in1, const float* __restrict__ in2,
                                int split, int Cin, int H, int W,
                                const uint32_t* __restrict__ Wp, int Cout,
                                const float* __restrict__ cb, const float* __restrict__ bg,
                                const float* __restrict__ bb, const float* __restrict__ bm,
                                const float* __restrict__ bv, float* __restrict__ out) {
  __shared__ alignas(16) _Float16 lA[16 * 32];
  const int tid  = threadIdx.x;
  const int wave = tid >> 5, lane = tid & 31;
  const int pix0 = blockIdx.x * 16;
  const int HW   = H * W;
  const int n    = pix0 / HW;
  const int rem  = pix0 - n * HW;
  const int y    = rem / W;
  const int x0   = rem - y * W;                 // multiple of 16 (W % 16 == 0)
  const int coutTile = blockIdx.y * 4 + wave;   // always valid (Cout in {64,128})
  const int nch = Cin >> 5;
  const int C2  = Cin - split;
  const int sm  = tid & 15;
  const int scl = (tid >> 4) * 4;
  const int nchunks = 9 * nch;
  v8f acc = {};
  for (int p = 0; p < 9; ++p) {
    const int dr = p / 3 - 1, ds = p % 3 - 1;
    const int yy = y + dr;
    const int xx = x0 + sm + ds;
    const bool inb = (yy >= 0) && (yy < H) && (xx >= 0) && (xx < W);
    for (int cbi = 0; cbi < nch; ++cbi) {
      union { _Float16 h[4]; uint64_t u; } pk;
      for (int i = 0; i < 4; ++i) {
        float v = 0.f;
        if (inb) {
          int cin = cbi * 32 + scl + i;
          v = (cin < split)
                ? in1[((size_t)(n * split + cin)) * HW + yy * W + xx]
                : in2[((size_t)(n * C2 + (cin - split))) * HW + yy * W + xx];
        }
        pk.h[i] = (_Float16)v;
      }
      *(uint64_t*)(&lA[sm * 32 + scl]) = pk.u;
      __syncthreads();
      {
        const int ml = lane & 15, g = lane >> 4;
        FragA a;
        a.h8[0] = *(const v8h*)(&lA[ml * 32 + g * 8]);
        a.h8[1] = *(const v8h*)(&lA[ml * 32 + 16 + g * 8]);
        const int c = p * nch + cbi;
        const uint32_t* bp = Wp + (((size_t)coutTile * nchunks + c) * 32 + lane) * 8;
        if (c + 1 < nchunks) __builtin_prefetch(bp + 256, 0, 1); // global_prefetch_b8
        FragB b;
        b.u4[0] = *(const v4u*)bp;
        b.u4[1] = *(const v4u*)(bp + 4);
        acc = __builtin_amdgcn_wmma_f32_16x16x32_f16(false, a.h16, false, b.h16,
                                                     (short)0, acc, false, false);
      }
      __syncthreads();
    }
  }
  const int g  = lane >> 4;
  const int co = coutTile * 16 + (lane & 15);
  const float scale = bg[co] * rsqrtf(bv[co] + 1e-5f);
  const float shift = (cb[co] - bm[co]) * scale + bb[co];
  float* op = out + ((size_t)(n * Cout + co)) * HW + y * W + x0;
  for (int v = 0; v < 8; ++v)
    op[v + g * 8] = fmaxf(acc[v] * scale + shift, 0.f);
}

// --------------------------- small VALU kernels -----------------------------

// First conv: Cin=3, direct, fused BN+ReLU.
__global__ void ms_conv3x3_c3(const float* __restrict__ x, const float* __restrict__ w,
                              const float* __restrict__ cb, const float* __restrict__ bg,
                              const float* __restrict__ bb, const float* __restrict__ bm,
                              const float* __restrict__ bv, float* __restrict__ out,
                              int NB, int Cout, int H, int W) {
  int idx = blockIdx.x * blockDim.x + threadIdx.x;
  int total = NB * Cout * H * W;
  if (idx >= total) return;
  int wq = idx % W, yq = (idx / W) % H, co = (idx / (W * H)) % Cout, n = idx / (W * H * Cout);
  float acc = 0.f;
  for (int c = 0; c < 3; ++c)
    for (int r = 0; r < 3; ++r)
      for (int s = 0; s < 3; ++s) {
        int yy = yq + r - 1, xx = wq + s - 1;
        if (yy >= 0 && yy < H && xx >= 0 && xx < W)
          acc += x[((size_t)(n * 3 + c) * H + yy) * W + xx] * w[((co * 3 + c) * 3 + r) * 3 + s];
      }
  float scale = bg[co] * rsqrtf(bv[co] + 1e-5f);
  float shift = (cb[co] - bm[co]) * scale + bb[co];
  out[idx] = fmaxf(acc * scale + shift, 0.f);
}

__global__ void ms_maxpool2(const float* __restrict__ in, float* __restrict__ out,
                            int NC, int H, int W) {
  int Ho = H >> 1, Wo = W >> 1;
  int idx = blockIdx.x * blockDim.x + threadIdx.x;
  int total = NC * Ho * Wo;
  if (idx >= total) return;
  int xq = idx % Wo, yq = (idx / Wo) % Ho, c = idx / (Wo * Ho);
  const float* p = in + (size_t)c * H * W + (2 * yq) * W + 2 * xq;
  out[idx] = fmaxf(fmaxf(p[0], p[1]), fmaxf(p[W], p[W + 1]));
}

__global__ void ms_patchify(const float* __restrict__ x4, float* __restrict__ P) {
  int idx = blockIdx.x * blockDim.x + threadIdx.x;   // [32][256][128]
  if (idx >= 32 * 256 * 128) return;
  int c = idx % 128, l = (idx / 128) % 256, seq = idx / (128 * 256);
  int n = seq >> 4, ph = (seq >> 2) & 3, pw = seq & 3;
  int py = l >> 4, px = l & 15;
  P[idx] = x4[(((size_t)n * 128 + c) * 64 + ph * 16 + py) * 64 + pw * 16 + px];
}

// causal depthwise conv1d (d_conv=4) + bias + SiLU; reads xs = xz[:, :256]
__global__ void ms_dwconv_silu(const float* __restrict__ xz, const float* __restrict__ cw,
                               const float* __restrict__ cbias, float* __restrict__ xsp,
                               int NS, int L) {
  int idx = blockIdx.x * blockDim.x + threadIdx.x;
  int total = NS * L * 256;
  if (idx >= total) return;
  int d = idx % 256, t = (idx / 256) % L, seq = idx / (256 * L);
  float acc = cbias[d];
  for (int k = 0; k < 4; ++k) {
    int tt = t - 3 + k;
    if (tt >= 0) acc += xz[((size_t)seq * L + tt) * 512 + d] * cw[d * 4 + k];
  }
  xsp[idx] = siluf(acc);
}

// dt = softplus(dbl[:, :8] @ dt_w^T + dt_b)
__global__ void ms_dtproj(const float* __restrict__ dbl, const float* __restrict__ dw,
                          const float* __restrict__ dbias, float* __restrict__ dtv, int M) {
  int idx = blockIdx.x * blockDim.x + threadIdx.x;
  if (idx >= M * 256) return;
  int m = idx / 256, d = idx % 256;
  float acc = dbias[d];
  for (int r = 0; r < 8; ++r) acc += dbl[(size_t)m * 48 + r] * dw[d * 8 + r];
  dtv[idx] = softplusf(acc);
}

// Selective scan; one block per sequence, thread = inner channel.
// Fuses D-skip and SiLU(z) gating. B/C staged in LDS (32 KB).
__global__ void ms_scan(const float* __restrict__ dbl, const float* __restrict__ dtv,
                        const float* __restrict__ xsp, const float* __restrict__ xz,
                        const float* __restrict__ Alog, const float* __restrict__ Dv,
                        float* __restrict__ yg, int L) {
  __shared__ float sB[256 * 16];
  __shared__ float sC[256 * 16];
  const int seq = blockIdx.x;
  const int d   = threadIdx.x;
  for (int i = d; i < L * 16; i += 256) {
    const size_t m = (size_t)seq * L + (i >> 4);
    sB[i] = dbl[m * 48 + 8  + (i & 15)];
    sC[i] = dbl[m * 48 + 24 + (i & 15)];
  }
  __syncthreads();
  float An[16], h[16];
  #pragma unroll
  for (int s = 0; s < 16; ++s) { An[s] = -__expf(Alog[d * 16 + s]); h[s] = 0.f; }
  const float Dd = Dv[d];
  for (int t = 0; t < L; ++t) {
    const size_t m  = (size_t)seq * L + t;
    const float dt = dtv[m * 256 + d];
    const float xv = xsp[m * 256 + d];
    const float dx = dt * xv;
    float a = 0.f;
    #pragma unroll
    for (int s = 0; s < 16; ++s) {
      h[s] = h[s] * __expf(dt * An[s]) + dx * sB[t * 16 + s];
      a += h[s] * sC[t * 16 + s];
    }
    const float z = xz[m * 512 + 256 + d];
    yg[m * 256 + d] = (a + xv * Dd) * siluf(z);
  }
}

__global__ void ms_unpatchify(const float* __restrict__ mo, float* __restrict__ mi) {
  int idx = blockIdx.x * blockDim.x + threadIdx.x;   // [2][128][64][64]
  if (idx >= 2 * 128 * 64 * 64) return;
  int wq = idx & 63, hq = (idx >> 6) & 63, c = (idx >> 12) & 127, n = idx >> 19;
  int ph = hq >> 4, py = hq & 15, pw = wq >> 4, px = wq & 15;
  int seq = n * 16 + ph * 4 + pw, l = py * 16 + px;
  mi[idx] = mo[((size_t)seq * 256 + l) * 128 + c];
}

// ConvTranspose2d k=2 s=2 (non-overlapping); w: [Cin, Cout, 2, 2]
__global__ void ms_upconv2(const float* __restrict__ in, const float* __restrict__ w,
                           const float* __restrict__ bias, float* __restrict__ out,
                           int NB, int Cin, int Cout, int H, int W) {
  int W2 = 2 * W, H2 = 2 * H;
  int idx = blockIdx.x * blockDim.x + threadIdx.x;
  int total = NB * Cout * H2 * W2;
  if (idx >= total) return;
  int ww = idx % W2, hh = (idx / W2) % H2, o = (idx / (W2 * H2)) % Cout, n = idx / (W2 * H2 * Cout);
  int y = hh >> 1, ky = hh & 1, xq = ww >> 1, kx = ww & 1;
  float acc = bias[o];
  for (int c = 0; c < Cin; ++c)
    acc += in[((size_t)(n * Cin + c) * H + y) * W + xq] * w[((c * Cout + o) * 2 + ky) * 2 + kx];
  out[idx] = acc;
}

// Fused concat(up2, x1) + 1x1 conv head; w2: [2, 128]
__global__ void ms_head(const float* __restrict__ up2, const float* __restrict__ x1,
                        const float* __restrict__ w2, const float* __restrict__ b2,
                        float* __restrict__ out, int NB, int H, int W) {
  int idx = blockIdx.x * blockDim.x + threadIdx.x;
  int total = NB * 2 * H * W;
  if (idx >= total) return;
  int xq = idx % W, yq = (idx / W) % H, o = (idx / (W * H)) % 2, n = idx / (W * H * 2);
  float acc = b2[o];
  for (int c = 0; c < 64; ++c)
    acc += up2[((size_t)(n * 64 + c) * H + yq) * W + xq] * w2[o * 128 + c];
  for (int c = 0; c < 64; ++c)
    acc += x1[((size_t)(n * 64 + c) * H + yq) * W + xq] * w2[o * 128 + 64 + c];
  out[idx] = acc;
}

// 3x3 avgpool, stride 1, pad 1, count_include_pad=True
__global__ void ms_avgpool3(const float* __restrict__ in, float* __restrict__ out,
                            int NC, int H, int W) {
  int idx = blockIdx.x * blockDim.x + threadIdx.x;
  if (idx >= NC * H * W) return;
  int xq = idx % W, yq = (idx / W) % H, c = idx / (W * H);
  const float* p = in + (size_t)c * H * W;
  float s = 0.f;
  for (int dy = -1; dy <= 1; ++dy)
    for (int dx = -1; dx <= 1; ++dx) {
      int yy = yq + dy, xx = xq + dx;
      if (yy >= 0 && yy < H && xx >= 0 && xx < W) s += p[yy * W + xx];
    }
  out[idx] = s * (1.f / 9.f);
}

// ---------------------------------------------------------------------------
extern "C" void kernel_launch(void* const* d_in, const int* in_sizes, int n_in,
                              void* d_out, int out_size, void* d_ws, size_t ws_size,
                              hipStream_t stream) {
  (void)in_sizes; (void)n_in; (void)out_size; (void)ws_size;
  const float* x      = (const float*)d_in[0];
  const float* e1c1w  = (const float*)d_in[1];  const float* e1c1b = (const float*)d_in[2];
  const float* e1c2w  = (const float*)d_in[3];  const float* e1c2b = (const float*)d_in[4];
  const float* e1g1   = (const float*)d_in[5];  const float* e1b1  = (const float*)d_in[6];
  const float* e1m1   = (const float*)d_in[7];  const float* e1v1  = (const float*)d_in[8];
  const float* e1g2   = (const float*)d_in[9];  const float* e1b2  = (const float*)d_in[10];
  const float* e1m2   = (const float*)d_in[11]; const float* e1v2  = (const float*)d_in[12];
  const float* e2c1w  = (const float*)d_in[13]; const float* e2c1b = (const float*)d_in[14];
  const float* e2c2w  = (const float*)d_in[15]; const float* e2c2b = (const float*)d_in[16];
  const float* e2g1   = (const float*)d_in[17]; const float* e2b1  = (const float*)d_in[18];
  const float* e2m1   = (const float*)d_in[19]; const float* e2v1  = (const float*)d_in[20];
  const float* e2g2   = (const float*)d_in[21]; const float* e2b2  = (const float*)d_in[22];
  const float* e2m2   = (const float*)d_in[23]; const float* e2v2  = (const float*)d_in[24];
  const float* d1c1w  = (const float*)d_in[25]; const float* d1c1b = (const float*)d_in[26];
  const float* d1c2w  = (const float*)d_in[27]; const float* d1c2b = (const float*)d_in[28];
  const float* d1g1   = (const float*)d_in[29]; const float* d1b1  = (const float*)d_in[30];
  const float* d1m1   = (const float*)d_in[31]; const float* d1v1  = (const float*)d_in[32];
  const float* d1g2   = (const float*)d_in[33]; const float* d1b2  = (const float*)d_in[34];
  const float* d1m2   = (const float*)d_in[35]; const float* d1v2  = (const float*)d_in[36];
  const float* up1w   = (const float*)d_in[37]; const float* up1bb = (const float*)d_in[38];
  const float* up2w   = (const float*)d_in[39]; const float* up2bb = (const float*)d_in[40];
  const float* d2w    = (const float*)d_in[41]; const float* d2b   = (const float*)d_in[42];
  const float* inprjw = (const float*)d_in[43];
  const float* c1dw   = (const float*)d_in[44]; const float* c1db  = (const float*)d_in[45];
  const float* xprjw  = (const float*)d_in[46];
  const float* dtw    = (const float*)d_in[47]; const float* dtbias= (const float*)d_in[48];
  const float* Alog   = (const float*)d_in[49]; const float* Dvec  = (const float*)d_in[50];
  const float* oprjw  = (const float*)d_in[51];
  float* outp = (float*)d_out;

  char* base = (char*)d_ws; size_t off = 0;
  auto alloc = [&](size_t bytes) -> void* {
    void* p = base + off; off = (off + bytes + 255) & ~(size_t)255; return p;
  };
  float* t1   = (float*)alloc((size_t)8388608 * 4);
  float* x1   = (float*)alloc((size_t)8388608 * 4);
  float* x2   = (float*)alloc((size_t)2097152 * 4);
  float* t3   = (float*)alloc((size_t)4194304 * 4);
  float* x3   = (float*)alloc((size_t)4194304 * 4);
  float* x4   = (float*)alloc((size_t)1048576 * 4);
  float* pat  = (float*)alloc((size_t)1048576 * 4);
  float* xzb  = (float*)alloc((size_t)4194304 * 4);
  float* xsp  = (float*)alloc((size_t)2097152 * 4);
  float* dbl  = (float*)alloc((size_t)393216  * 4);
  float* dtb  = (float*)alloc((size_t)2097152 * 4);
  float* ygb  = (float*)alloc((size_t)2097152 * 4);
  float* mo   = (float*)alloc((size_t)1048576 * 4);
  float* mi   = (float*)alloc((size_t)1048576 * 4);
  float* up1o = (float*)alloc((size_t)4194304 * 4);
  float* d1t  = (float*)alloc((size_t)2097152 * 4);
  float* d1o  = (float*)alloc((size_t)2097152 * 4);
  float* up2o = (float*)alloc((size_t)8388608 * 4);
  float* d2o  = (float*)alloc((size_t)262144  * 4);
  uint32_t* pwE1c2 = (uint32_t*)alloc((size_t)18432 * 4);
  uint32_t* pwE2c1 = (uint32_t*)alloc((size_t)36864 * 4);
  uint32_t* pwE2c2 = (uint32_t*)alloc((size_t)73728 * 4);
  uint32_t* pwD1c1 = (uint32_t*)alloc((size_t)73728 * 4);
  uint32_t* pwD1c2 = (uint32_t*)alloc((size_t)18432 * 4);
  uint32_t* pwInP  = (uint32_t*)alloc((size_t)32768 * 4);
  uint32_t* pwXP   = (uint32_t*)alloc((size_t)6144  * 4);
  uint32_t* pwOutP = (uint32_t*)alloc((size_t)16384 * 4);

  auto cdiv = [](int a, int b) { return (a + b - 1) / b; };
  // ---- pack all WMMA weights (conv mode=1, gemm mode=0) ----
  ms_pack_b<<<cdiv(18432,256),256,0,stream>>>(e1c2w, pwE1c2, 18432, 18, 576,  64, 1, 64);
  ms_pack_b<<<cdiv(36864,256),256,0,stream>>>(e2c1w, pwE2c1, 36864, 18, 576, 128, 1, 64);
  ms_pack_b<<<cdiv(73728,256),256,0,stream>>>(e2c2w, pwE2c2, 73728, 36, 1152,128, 1, 128);
  ms_pack_b<<<cdiv(73728,256),256,0,stream>>>(d1c1w, pwD1c1, 73728, 72, 2304, 64, 1, 256);
  ms_pack_b<<<cdiv(18432,256),256,0,stream>>>(d1c2w, pwD1c2, 18432, 18, 576,  64, 1, 64);
  ms_pack_b<<<cdiv(32768,256),256,0,stream>>>(inprjw,pwInP,  32768,  4, 128, 512, 0, 0);
  ms_pack_b<<<cdiv(6144 ,256),256,0,stream>>>(xprjw, pwXP,    6144,  8, 256,  40, 0, 0);
  ms_pack_b<<<cdiv(16384,256),256,0,stream>>>(oprjw, pwOutP, 16384,  8, 256, 128, 0, 0);

  // ---- encoder ----
  ms_conv3x3_c3<<<cdiv(8388608,256),256,0,stream>>>(x, e1c1w, e1c1b, e1g1, e1b1, e1m1, e1v1,
                                                    t1, 2, 64, 256, 256);
  ms_conv3x3_wmma<<<dim3(8192,1),128,0,stream>>>(t1, t1, 64, 64, 256, 256, pwE1c2, 64,
                                                 e1c2b, e1g2, e1b2, e1m2, e1v2, x1);
  ms_maxpool2<<<cdiv(2097152,256),256,0,stream>>>(x1, x2, 128, 256, 256);
  ms_conv3x3_wmma<<<dim3(2048,2),128,0,stream>>>(x2, x2, 64, 64, 128, 128, pwE2c1, 128,
                                                 e2c1b, e2g1, e2b1, e2m1, e2v1, t3);
  ms_conv3x3_wmma<<<dim3(2048,2),128,0,stream>>>(t3, t3, 128, 128, 128, 128, pwE2c2, 128,
                                                 e2c2b, e2g2, e2b2, e2m2, e2v2, x3);
  ms_maxpool2<<<cdiv(1048576,256),256,0,stream>>>(x3, x4, 256, 128, 128);

  // ---- mamba on 16x16 patches (32 seqs x L=256 x d=128) ----
  ms_patchify<<<cdiv(1048576,256),256,0,stream>>>(x4, pat);
  ms_gemm_wmma<<<dim3(512,8),128,0,stream>>>(pat, pwInP, xzb, 128, 512);
  ms_dwconv_silu<<<cdiv(2097152,256),256,0,stream>>>(xzb, c1dw, c1db, xsp, 32, 256);
  ms_gemm_wmma<<<dim3(512,1),128,0,stream>>>(xsp, pwXP, dbl, 256, 48);
  ms_dtproj<<<cdiv(2097152,256),256,0,stream>>>(dbl, dtw, dtbias, dtb, 8192);
  ms_scan<<<32,256,0,stream>>>(dbl, dtb, xsp, xzb, Alog, Dvec, ygb, 256);
  ms_gemm_wmma<<<dim3(512,2),128,0,stream>>>(ygb, pwOutP, mo, 256, 128);
  ms_unpatchify<<<cdiv(1048576,256),256,0,stream>>>(mo, mi);

  // ---- decoder ----
  ms_upconv2<<<cdiv(4194304,256),256,0,stream>>>(mi, up1w, up1bb, up1o, 2, 128, 128, 64, 64);
  ms_conv3x3_wmma<<<dim3(2048,1),128,0,stream>>>(up1o, x3, 128, 256, 128, 128, pwD1c1, 64,
                                                 d1c1b, d1g1, d1b1, d1m1, d1v1, d1t);
  ms_conv3x3_wmma<<<dim3(2048,1),128,0,stream>>>(d1t, d1t, 64, 64, 128, 128, pwD1c2, 64,
                                                 d1c2b, d1g2, d1b2, d1m2, d1v2, d1o);
  ms_upconv2<<<cdiv(8388608,256),256,0,stream>>>(d1o, up2w, up2bb, up2o, 2, 64, 64, 128, 128);
  ms_head<<<cdiv(262144,256),256,0,stream>>>(up2o, x1, d2w, d2b, d2o, 2, 256, 256);
  ms_avgpool3<<<cdiv(262144,256),256,0,stream>>>(d2o, outp, 4, 256, 256);
}
